// clip_image_encoder_56547539419751
// MI455X (gfx1250) — matrix operand order repaired and checked by reading, not compiled, not run on hardware
//
#include <hip/hip_runtime.h>
#include <hip/hip_bf16.h>
#include <stdint.h>

typedef __attribute__((ext_vector_type(16))) _Float16 v16h;
typedef __attribute__((ext_vector_type(8)))  _Float16 v8h;
typedef __attribute__((ext_vector_type(8)))  float    v8f;

#define CAT16(lo, hi) __builtin_shufflevector((lo), (hi), 0,1,2,3,4,5,6,7,8,9,10,11,12,13,14,15)

__device__ __forceinline__ v8f wmma16(v16h a, v16h b, v8f c) {
  // D = A(16x32 f16) * B(32x16 f16) + C(16x16 f32)
  return __builtin_amdgcn_wmma_f32_16x16x32_f16(false, a, false, b, (short)0, c, false, false);
}

// async 16B copy global -> LDS (GLOBAL_LOAD_ASYNC_TO_LDS_B128, ASYNCcnt-tracked).
// LDS byte offset = low 32 bits of the generic shared-aperture pointer (ISA 10.2).
__device__ __forceinline__ void async_copy_b128(void* lds_ptr, const void* gptr) {
  unsigned lds = (unsigned)(uintptr_t)lds_ptr;
  asm volatile("global_load_async_to_lds_b128 %0, %1, off"
               :: "v"(lds), "v"(gptr) : "memory");
}
__device__ __forceinline__ void async_wait0() {
  asm volatile("s_wait_asynccnt 0x0" ::: "memory");
}

// ---------------------------------------------------------------------------
// Conversion / packing kernels
// ---------------------------------------------------------------------------
__global__ void convert_f32_f16(const float* __restrict__ src, _Float16* __restrict__ dst, long n) {
  long i = (long)blockIdx.x * blockDim.x + threadIdx.x;
  if (i < n) dst[i] = (_Float16)src[i];
}

// dst[c*R + r] = src[r*C + c]   (weight [R,C] row-major -> B matrix [C,R] = W^T)
__global__ void transpose_f32_f16(const float* __restrict__ src, _Float16* __restrict__ dst,
                                  int R, int C) {
  long i = (long)blockIdx.x * blockDim.x + threadIdx.x;
  if (i < (long)R * C) {
    int r = (int)(i / C);
    int c = (int)(i % C);
    dst[(long)c * R + r] = (_Float16)src[i];
  }
}

// ---------------------------------------------------------------------------
// Generic WMMA GEMM: C[M,N] = act(A_f16[M,K] @ B_f16[K,N] + bias (+ resid))
// Block tile 128x128, K-step 32, 8 waves, wave tile 32x64.
// A tile staged via async global->LDS DMA; B tile transposed through VGPRs.
// ---------------------------------------------------------------------------
#define BM 128
#define BN 128
#define BK 32
#define AS_STRIDE 40   // 32 + 8 pad (halfs)
#define BS_STRIDE 40

template<bool RELU, bool HAS_RES, bool OUT32, bool OUT16>
__global__ __launch_bounds__(256)
void gemm_f16_wmma(const _Float16* __restrict__ A, const _Float16* __restrict__ B,
                   const float* __restrict__ bias, const float* __restrict__ resid,
                   float* __restrict__ C32, _Float16* __restrict__ C16,
                   int M, int N, int K) {
  __shared__ _Float16 As[BM * AS_STRIDE];
  __shared__ _Float16 Bs[BN * BS_STRIDE];
  const int tid   = threadIdx.x;
  const int lane  = tid & 31;
  const int wave  = tid >> 5;
  const int wm    = wave & 3;   // 4 waves along M
  const int wn    = wave >> 2;  // 2 waves along N
  const int ln16  = lane & 15;
  const int lhalf = lane >> 4;  // 0 | 1
  const int m0 = blockIdx.y * BM;
  const int n0 = blockIdx.x * BN;

  v8f acc[2][4];
  const v8f vz = {0.f,0.f,0.f,0.f,0.f,0.f,0.f,0.f};
#pragma unroll
  for (int i = 0; i < 2; ++i)
#pragma unroll
    for (int j = 0; j < 4; ++j) acc[i][j] = vz;

  for (int k0 = 0; k0 < K; k0 += BK) {
    // stage A tile (row-major copy): async DMA, 512 16B chunks
#pragma unroll
    for (int i = 0; i < 2; ++i) {
      int chunk = tid + i * 256;
      int row = chunk >> 2;
      int cc  = (chunk & 3) * 8;
      async_copy_b128(As + row * AS_STRIDE + cc,
                      A + (long)(m0 + row) * K + k0 + cc);
      if (k0 + BK < K)
        __builtin_prefetch(A + (long)(m0 + row) * K + k0 + BK + cc, 0, 1);
    }
    // stage B tile transposed (N-major): Bs[n][k]
#pragma unroll
    for (int i = 0; i < 2; ++i) {
      int chunk = tid + i * 256;
      int kr = chunk >> 4;          // 0..31
      int nc = (chunk & 15) * 8;    // 0..120
      v8h v = *(const v8h*)(B + (long)(k0 + kr) * N + n0 + nc);
#pragma unroll
      for (int j = 0; j < 8; ++j) Bs[(nc + j) * BS_STRIDE + kr] = v[j];
      if (k0 + BK < K)
        __builtin_prefetch(B + (long)(k0 + BK + kr) * N + n0 + nc, 0, 1);
    }
    async_wait0();
    __syncthreads();

    v16h af[2];
#pragma unroll
    for (int sm = 0; sm < 2; ++sm) {
      const _Float16* p = As + (wm * 32 + sm * 16 + ln16) * AS_STRIDE + lhalf * 8;
      v8h lo = *(const v8h*)(p);
      v8h hi = *(const v8h*)(p + 16);
      af[sm] = CAT16(lo, hi);
    }
#pragma unroll
    for (int sn = 0; sn < 4; ++sn) {
      const _Float16* p = Bs + (wn * 64 + sn * 16 + ln16) * BS_STRIDE + lhalf * 8;
      v8h lo = *(const v8h*)(p);
      v8h hi = *(const v8h*)(p + 16);
      v16h bf = CAT16(lo, hi);
      acc[0][sn] = wmma16(af[0], bf, acc[0][sn]);
      acc[1][sn] = wmma16(af[1], bf, acc[1][sn]);
    }
    __syncthreads();
  }

  // branch-free epilogue (all feature selection is compile-time)
#pragma unroll
  for (int sm = 0; sm < 2; ++sm) {
#pragma unroll
    for (int sn = 0; sn < 4; ++sn) {
      const int col = n0 + wn * 64 + sn * 16 + ln16;
      const float bv = bias[col];
      const int rbase = m0 + wm * 32 + sm * 16 + lhalf * 8;
#pragma unroll
      for (int r = 0; r < 8; ++r) {
        const long idx = (long)(rbase + r) * N + col;
        float x = acc[sm][sn][r] + bv;
        if constexpr (RELU)    x = fmaxf(x, 0.f);
        if constexpr (HAS_RES) x += resid[idx];
        if constexpr (OUT32)   C32[idx] = x;
        if constexpr (OUT16)   C16[idx] = (_Float16)x;
      }
    }
  }
}

// ---------------------------------------------------------------------------
// Flash attention: per block (b,h, 128 query rows); wave handles 16 q-rows.
// QKV row layout per token: [q(512) | k(512) | v(512)], head h = dims h*128..
// ---------------------------------------------------------------------------
#define KS_STRIDE 136  // 128 + 8
#define VS_STRIDE 40   // 32 + 8
#define PS_STRIDE 40

__global__ __launch_bounds__(256)
void attention_wmma(const _Float16* __restrict__ QKV, _Float16* __restrict__ O, int S) {
  __shared__ _Float16 Ks[32 * KS_STRIDE];    // [key t][dim d] row-major
  __shared__ _Float16 Vs[128 * VS_STRIDE];   // [dim d][key t] (transposed)
  __shared__ _Float16 Ps[8 * 16 * PS_STRIDE];
  const int bh   = blockIdx.x;        // b*4 + h
  const int b    = bh >> 2;
  const int h    = bh & 3;
  const int qblk = blockIdx.y;        // S/128 blocks
  const int tid  = threadIdx.x;
  const int lane = tid & 31;
  const int wave = tid >> 5;
  const int ln16 = lane & 15;
  const int lhalf = lane >> 4;
  const long base = (long)b * S * 1536;
  const float scale = 0.08838834764831845f;   // 1/sqrt(128)

  // Q fragments (A layout), HD=128 as 4 K-steps of 32, held in registers
  v16h qf[4];
  {
    const int qrow = qblk * 128 + wave * 16 + ln16;
    const _Float16* qp = QKV + base + (long)qrow * 1536 + h * 128;
#pragma unroll
    for (int f = 0; f < 4; ++f) {
      v8h lo = *(const v8h*)(qp + f * 32 + lhalf * 8);
      v8h hi = *(const v8h*)(qp + f * 32 + lhalf * 8 + 16);
      qf[f] = CAT16(lo, hi);
    }
  }

  v8f oacc[8];
  const v8f vz = {0.f,0.f,0.f,0.f,0.f,0.f,0.f,0.f};
#pragma unroll
  for (int i = 0; i < 8; ++i) oacc[i] = vz;
  float mrow[8], lrow[8];
#pragma unroll
  for (int r = 0; r < 8; ++r) { mrow[r] = -1e30f; lrow[r] = 0.f; }

  for (int kb = 0; kb < S / 32; ++kb) {
    // K tile 32x128 row-major: async global->LDS DMA (direct copy)
#pragma unroll
    for (int i = 0; i < 2; ++i) {
      int chunk = tid + i * 256;
      int t  = chunk >> 4;
      int dc = (chunk & 15) * 8;
      async_copy_b128(Ks + t * KS_STRIDE + dc,
                      QKV + base + (long)(kb * 32 + t) * 1536 + 512 + h * 128 + dc);
    }
    // V tile transposed: Vs[d][t]
#pragma unroll
    for (int i = 0; i < 2; ++i) {
      int chunk = tid + i * 256;
      int t  = chunk >> 4;
      int dc = (chunk & 15) * 8;
      v8h v = *(const v8h*)(QKV + base + (long)(kb * 32 + t) * 1536 + 1024 + h * 128 + dc);
#pragma unroll
      for (int j = 0; j < 8; ++j) Vs[(dc + j) * VS_STRIDE + t] = v[j];
    }
    async_wait0();
    __syncthreads();

    // scores: S_tile = Q(16x128) @ K^T -> two 16x16 f32 tiles
    v8f sc[2];
#pragma unroll
    for (int st = 0; st < 2; ++st) {
      sc[st] = vz;
#pragma unroll
      for (int f = 0; f < 4; ++f) {
        const _Float16* p = Ks + (st * 16 + ln16) * KS_STRIDE + f * 32 + lhalf * 8;
        v8h lo = *(const v8h*)(p);
        v8h hi = *(const v8h*)(p + 16);
        sc[st] = wmma16(qf[f], CAT16(lo, hi), sc[st]);
      }
    }
    // online softmax (rows split: lanes 0-15 own rows 0-7, lanes 16-31 rows 8-15)
    float tmax[8];
#pragma unroll
    for (int r = 0; r < 8; ++r) {
      float a = sc[0][r] * scale, c = sc[1][r] * scale;
      sc[0][r] = a; sc[1][r] = c;
      tmax[r] = fmaxf(a, c);
    }
#pragma unroll
    for (int off = 1; off < 16; off <<= 1)
#pragma unroll
      for (int r = 0; r < 8; ++r)
        tmax[r] = fmaxf(tmax[r], __shfl_xor(tmax[r], off, 32));
    float rs[8];
#pragma unroll
    for (int r = 0; r < 8; ++r) {
      float mn = fmaxf(mrow[r], tmax[r]);
      float cs = __expf(mrow[r] - mn);
      mrow[r] = mn;
      float p0 = __expf(sc[0][r] - mn);
      float p1 = __expf(sc[1][r] - mn);
      sc[0][r] = p0; sc[1][r] = p1;
      lrow[r] = lrow[r] * cs;
      rs[r] = p0 + p1;
#pragma unroll
      for (int dt = 0; dt < 8; ++dt) oacc[dt][r] *= cs;
    }
#pragma unroll
    for (int off = 1; off < 16; off <<= 1)
#pragma unroll
      for (int r = 0; r < 8; ++r)
        rs[r] += __shfl_xor(rs[r], off, 32);
#pragma unroll
    for (int r = 0; r < 8; ++r) lrow[r] += rs[r];

    // bounce P (C-layout f32) through LDS into A-layout f16
    _Float16* pw = Ps + wave * 16 * PS_STRIDE;
#pragma unroll
    for (int r = 0; r < 8; ++r) {
      int prow = r + lhalf * 8;
      pw[prow * PS_STRIDE + ln16]      = (_Float16)sc[0][r];
      pw[prow * PS_STRIDE + 16 + ln16] = (_Float16)sc[1][r];
    }
    asm volatile("s_wait_dscnt 0x0" ::: "memory");
    v16h pa;
    {
      const _Float16* p = pw + ln16 * PS_STRIDE + lhalf * 8;
      v8h lo = *(const v8h*)(p);
      v8h hi = *(const v8h*)(p + 16);
      pa = CAT16(lo, hi);
    }
    // O += P(16x32) @ V(32x128)
#pragma unroll
    for (int dt = 0; dt < 8; ++dt) {
      const _Float16* p = Vs + (dt * 16 + ln16) * VS_STRIDE + lhalf * 8;
      v8h lo = *(const v8h*)(p);
      v8h hi = *(const v8h*)(p + 16);
      oacc[dt] = wmma16(pa, CAT16(lo, hi), oacc[dt]);
    }
    __syncthreads();
  }

  // normalize + store (f16 for the out_proj GEMM)
#pragma unroll
  for (int dt = 0; dt < 8; ++dt) {
#pragma unroll
    for (int r = 0; r < 8; ++r) {
      int row = qblk * 128 + wave * 16 + r + lhalf * 8;
      float v = oacc[dt][r] / lrow[r];
      O[((long)(b * S + row)) * 512 + h * 128 + dt * 16 + ln16] = (_Float16)v;
    }
  }
}

// ---------------------------------------------------------------------------
// LayerNorm over last dim (512); one wave per row; optional f16 mirror.
// ---------------------------------------------------------------------------
__global__ __launch_bounds__(256)
void layernorm512(const float* __restrict__ X, const float* __restrict__ g,
                  const float* __restrict__ bta, float* __restrict__ Y32,
                  _Float16* __restrict__ Y16) {
  const int row  = blockIdx.x * 8 + (threadIdx.x >> 5);
  const int lane = threadIdx.x & 31;
  const float* xr = X + (long)row * 512;
  float x[16], s = 0.f, s2 = 0.f;
#pragma unroll
  for (int i = 0; i < 16; ++i) {
    x[i] = xr[i * 32 + lane];
    s += x[i]; s2 += x[i] * x[i];
  }
#pragma unroll
  for (int off = 1; off < 32; off <<= 1) {
    s  += __shfl_xor(s,  off, 32);
    s2 += __shfl_xor(s2, off, 32);
  }
  float mean = s * (1.f / 512.f);
  float var  = s2 * (1.f / 512.f) - mean * mean;
  float rstd = rsqrtf(var + 1e-5f);
#pragma unroll
  for (int i = 0; i < 16; ++i) {
    int c = i * 32 + lane;
    float y = (x[i] - mean) * rstd * g[c] + bta[c];
    Y32[(long)row * 512 + c] = y;
    if (Y16) Y16[(long)row * 512 + c] = (_Float16)y;
  }
}

// ---------------------------------------------------------------------------
// Masked mean pool over ragged prefix + 2-class head. One block per batch b.
// ---------------------------------------------------------------------------
__global__ __launch_bounds__(256)
void pool_cls(const float* __restrict__ Z, const int* __restrict__ mask,
              const float* __restrict__ wcls, const float* __restrict__ bcls,
              float* __restrict__ out, int S) {
  __shared__ float red0[256];
  __shared__ float red1[256];
  const int b = blockIdx.x;
  const int t = threadIdx.x;
  float a0 = 0.f, a1 = 0.f, cnt = 0.f;
  for (int s = 0; s < S; ++s) {
    float m = (mask[b * S + s] > 0) ? 1.f : 0.f;
    cnt += m;
    const float* zr = Z + ((long)(b * S + s)) * 512;
    a0 += m * zr[t];
    a1 += m * zr[t + 256];
  }
  float inv = 1.f / cnt;
  float p0 = a0 * inv, p1 = a1 * inv;
  red0[t] = p0 * wcls[t] + p1 * wcls[t + 256];
  red1[t] = p0 * wcls[512 + t] + p1 * wcls[512 + t + 256];
  __syncthreads();
  for (int off = 128; off > 0; off >>= 1) {
    if (t < off) { red0[t] += red0[t + off]; red1[t] += red1[t + off]; }
    __syncthreads();
  }
  if (t == 0) {
    out[b * 2 + 0] = red0[0] + bcls[0];
    out[b * 2 + 1] = red1[0] + bcls[1];
  }
}

// ---------------------------------------------------------------------------
extern "C" void kernel_launch(void* const* d_in, const int* in_sizes, int n_in,
                              void* d_out, int out_size, void* d_ws, size_t ws_size,
                              hipStream_t stream) {
  (void)in_sizes; (void)n_in; (void)out_size; (void)ws_size;
  const int B = 64, S = 512, M = B * S;      // 32768 tokens
  const int DIN = 768, D = 512, DFF = 2048, DQKV = 1536;

  const float* vf         = (const float*)d_in[0];
  const int*   vmask      = (const int*)d_in[1];
  const float* w_img      = (const float*)d_in[2];
  const float* b_img      = (const float*)d_in[3];
  const float* in_proj_w  = (const float*)d_in[4];
  const float* in_proj_b  = (const float*)d_in[5];
  const float* out_proj_w = (const float*)d_in[6];
  const float* out_proj_b = (const float*)d_in[7];
  const float* ln1_g = (const float*)d_in[8];
  const float* ln1_b = (const float*)d_in[9];
  const float* w1    = (const float*)d_in[10];
  const float* b1    = (const float*)d_in[11];
  const float* w2    = (const float*)d_in[12];
  const float* b2    = (const float*)d_in[13];
  const float* ln2_g = (const float*)d_in[14];
  const float* ln2_b = (const float*)d_in[15];
  const float* w_cls = (const float*)d_in[16];
  const float* b_cls = (const float*)d_in[17];
  float* out = (float*)d_out;

  // workspace carve-up (256B aligned)
  char* ws = (char*)d_ws;
  size_t off = 0;
  auto alloc = [&](size_t bytes) -> void* {
    void* p = ws + off;
    off = (off + bytes + 255) & ~(size_t)255;
    return p;
  };
  _Float16* wimg_t   = (_Float16*)alloc((size_t)DIN * D * 2);
  _Float16* winp_t   = (_Float16*)alloc((size_t)D * DQKV * 2);
  _Float16* wout_t   = (_Float16*)alloc((size_t)D * D * 2);
  _Float16* w1_t     = (_Float16*)alloc((size_t)D * DFF * 2);
  _Float16* w2_t     = (_Float16*)alloc((size_t)DFF * D * 2);
  _Float16* vf16     = (_Float16*)alloc((size_t)M * DIN * 2);
  float*    X32      = (float*)   alloc((size_t)M * D * 4);   // x, later y, later z
  _Float16* X16      = (_Float16*)alloc((size_t)M * D * 2);   // f16 mirror
  _Float16* QKV16    = (_Float16*)alloc((size_t)M * DQKV * 2);
  _Float16* O16      = (_Float16*)alloc((size_t)M * D * 2);
  float*    R32      = (float*)   alloc((size_t)M * D * 4);   // pre-LN residual sum
  _Float16* FF116    = (_Float16*)alloc((size_t)M * DFF * 2);

  // pack weights (transpose to [K,N] f16)
  auto tgrid = [](long n) { return dim3((unsigned)((n + 255) / 256)); };
  transpose_f32_f16<<<tgrid((long)D * DIN),   256, 0, stream>>>(w_img,      wimg_t, D,    DIN);
  transpose_f32_f16<<<tgrid((long)DQKV * D),  256, 0, stream>>>(in_proj_w,  winp_t, DQKV, D);
  transpose_f32_f16<<<tgrid((long)D * D),     256, 0, stream>>>(out_proj_w, wout_t, D,    D);
  transpose_f32_f16<<<tgrid((long)DFF * D),   256, 0, stream>>>(w1,         w1_t,   DFF,  D);
  transpose_f32_f16<<<tgrid((long)D * DFF),   256, 0, stream>>>(w2,         w2_t,   D,    DFF);
  convert_f32_f16  <<<tgrid((long)M * DIN),   256, 0, stream>>>(vf, vf16, (long)M * DIN);

  // 1) x = relu(vf @ Wimg^T + b)  -> f32 + f16
  gemm_f16_wmma<true, false, true, true><<<dim3(D / BN, M / BM), 256, 0, stream>>>(
      vf16, wimg_t, b_img, nullptr, X32, X16, M, D, DIN);
  // 2) qkv = x @ Win^T + b  -> f16
  gemm_f16_wmma<false, false, false, true><<<dim3(DQKV / BN, M / BM), 256, 0, stream>>>(
      X16, winp_t, in_proj_b, nullptr, nullptr, QKV16, M, DQKV, D);
  // 3) flash attention -> O f16
  attention_wmma<<<dim3(B * 4, S / 128), 256, 0, stream>>>(QKV16, O16, S);
  // 4) o2 = O @ Wout^T + b + x(residual) -> R32
  gemm_f16_wmma<false, true, true, false><<<dim3(D / BN, M / BM), 256, 0, stream>>>(
      O16, wout_t, out_proj_b, X32, R32, nullptr, M, D, D);
  // 5) LN1: R32 -> y (reuse X32/X16)
  layernorm512<<<dim3(M / 8), 256, 0, stream>>>(R32, ln1_g, ln1_b, X32, X16);
  // 6) ff1 = relu(y @ W1^T + b1) -> f16
  gemm_f16_wmma<true, false, false, true><<<dim3(DFF / BN, M / BM), 256, 0, stream>>>(
      X16, w1_t, b1, nullptr, nullptr, FF116, M, DFF, D);
  // 7) ff2 = ff1 @ W2^T + b2 + y(residual) -> R32
  gemm_f16_wmma<false, true, true, false><<<dim3(D / BN, M / BM), 256, 0, stream>>>(
      FF116, w2_t, b2, X32, R32, nullptr, M, D, DFF);
  // 8) LN2: R32 -> z (reuse X32)
  layernorm512<<<dim3(M / 8), 256, 0, stream>>>(R32, ln2_g, ln2_b, X32, nullptr);
  // 9) masked mean pool + classifier head
  pool_cls<<<dim3(B), 256, 0, stream>>>(X32, vmask, w_cls, b_cls, out, S);
}
